// FineTuneNarrativeModel_15874199126542
// MI455X (gfx1250) — matrix operand order repaired and checked by reading, not compiled, not run on hardware
//
#include <hip/hip_runtime.h>
#include <hip/hip_bf16.h>

typedef __attribute__((ext_vector_type(16))) _Float16 v16h;
typedef __attribute__((ext_vector_type(8)))  _Float16 v8h;
typedef __attribute__((ext_vector_type(8)))  float    v8f;
typedef __attribute__((ext_vector_type(4)))  float    v4f;

#define BB   16
#define SS   1024
#define CC   256
#define HH   4
#define DHD  64
#define DFFD 2048
#define NEGV (-1e9f)

// ---------------------------------------------------------------------------
// WMMA fragment helpers (CDNA5 16x16x32 f16 layouts, ISA 7.12.2)
// A 16x32: lane<16 holds K {0..7, 16..23}; lane>=16 holds K {8..15, 24..31}
// B 32x16: lane<16 holds K {0..7, 16..23}; lane>=16 holds K {8..15, 24..31}
// ---------------------------------------------------------------------------
__device__ __forceinline__ v16h zero_v16h() {
  v16h r;
#pragma unroll
  for (int i = 0; i < 16; ++i) r[i] = (_Float16)0.0f;
  return r;
}

__device__ __forceinline__ v16h load_a_f16(const _Float16* base) {
  v8h c0 = *(const v8h*)(base);
  v8h c1 = *(const v8h*)(base + 16);
  v16h r;
#pragma unroll
  for (int i = 0; i < 8; ++i) { r[i] = c0[i]; r[i + 8] = c1[i]; }
  return r;
}

__device__ __forceinline__ v16h load_a_f32(const float* base) {
  v4f c0 = *(const v4f*)(base);
  v4f c1 = *(const v4f*)(base + 4);
  v4f c2 = *(const v4f*)(base + 16);
  v4f c3 = *(const v4f*)(base + 20);
  v16h r;
#pragma unroll
  for (int i = 0; i < 4; ++i) {
    r[i]      = (_Float16)c0[i];
    r[i + 4]  = (_Float16)c1[i];
    r[i + 8]  = (_Float16)c2[i];
    r[i + 12] = (_Float16)c3[i];
  }
  return r;
}

__device__ __forceinline__ v16h load_b_f16(const _Float16* base) {
  v8h c0 = *(const v8h*)(base);
  v8h c1 = *(const v8h*)(base + 16);
  v16h r;
#pragma unroll
  for (int i = 0; i < 8; ++i) { r[i] = c0[i]; r[i + 8] = c1[i]; }
  return r;
}

// ---------------------------------------------------------------------------
// Generic activation @ W.T GEMM.  W stored [N, K] row-major (as in reference).
// Optional conv-tap accumulation (tap-major packed weights, causal row shift).
// Epilogue modes: 0=none, 1=relu, 2=relu->+res->relu (TCN block), 3=+res.
// One wave = 64x64 output tile. 4 waves / block cover 4 N-tiles.
// ---------------------------------------------------------------------------
struct GemmWSParams {
  const _Float16* A;   int lda;
  const _Float16* W;   int ldw;
  const float*    bias;
  const float*    res; int ldres;
  float*          out32; int ldo32;
  _Float16*       out16; int ldo16;
  int M, N, K;
  int taps, dil, seqMask;  // seqMask = SS-1 for conv, else 0
  int mode;
};

__global__ __launch_bounds__(128) void gemm_ws_kernel(GemmWSParams p) {
  const int lane  = threadIdx.x & 31;
  const int wave  = threadIdx.x >> 5;
  const int nTiles = p.N >> 6;
  const int ntile = blockIdx.x * 4 + wave;
  if (ntile >= nTiles) return;
  const int tm = blockIdx.y << 6;
  const int tn = ntile << 6;
  const int l  = lane & 15;
  const int hi = lane >> 4;
  const int aOff = hi ? 8 : 0;
  const int bOff = hi ? 8 : 0;

  v8f acc[4][4];
#pragma unroll
  for (int a = 0; a < 4; ++a)
#pragma unroll
    for (int b = 0; b < 4; ++b)
#pragma unroll
      for (int r = 0; r < 8; ++r) acc[a][b][r] = 0.0f;

  const size_t wTap = (size_t)p.N * p.ldw;
  for (int t = 0; t < p.taps; ++t) {
    const int shift = (p.taps - 1 - t) * p.dil;
    const _Float16* Wt = p.W + (size_t)t * wTap;
    for (int k0 = 0; k0 < p.K; k0 += 32) {
      v16h af[4], bf[4];
#pragma unroll
      for (int mi = 0; mi < 4; ++mi) {
        const int row = tm + mi * 16 + l;
        if ((row & p.seqMask) >= shift)
          af[mi] = load_a_f16(p.A + (size_t)(row - shift) * p.lda + k0 + aOff);
        else
          af[mi] = zero_v16h();
      }
#pragma unroll
      for (int ni = 0; ni < 4; ++ni)
        bf[ni] = load_b_f16(Wt + (size_t)(tn + ni * 16 + l) * p.ldw + k0 + bOff);
#pragma unroll
      for (int mi = 0; mi < 4; ++mi)
#pragma unroll
        for (int ni = 0; ni < 4; ++ni)
          acc[mi][ni] = __builtin_amdgcn_wmma_f32_16x16x32_f16(
              false, af[mi], false, bf[ni], (short)0, acc[mi][ni], false, false);
    }
  }

  const int mBase = hi ? 8 : 0;
#pragma unroll
  for (int ni = 0; ni < 4; ++ni) {
    const int n = tn + ni * 16 + l;
    const float bv = p.bias ? p.bias[n] : 0.0f;
#pragma unroll
    for (int mi = 0; mi < 4; ++mi) {
#pragma unroll
      for (int r = 0; r < 8; ++r) {
        const int m = tm + mi * 16 + mBase + r;
        float v = acc[mi][ni][r] + bv;
        if (p.mode == 1)      v = fmaxf(v, 0.0f);
        else if (p.mode == 2) v = fmaxf(fmaxf(v, 0.0f) + p.res[(size_t)m * p.ldres + n], 0.0f);
        else if (p.mode == 3) v += p.res[(size_t)m * p.ldres + n];
        if (p.out32) p.out32[(size_t)m * p.ldo32 + n] = v;
        if (p.out16) p.out16[(size_t)m * p.ldo16 + n] = (_Float16)v;
      }
    }
  }
}

// ---------------------------------------------------------------------------
// scores = (Q @ K^T) / 8, key-pad mask -> -1e9, written f32 into attn buffer.
// Batched over (b,h) via blockIdx.z. Q/K rows live in the f16 qkv buffer.
// ---------------------------------------------------------------------------
__global__ __launch_bounds__(128) void gemm_scores_kernel(
    const _Float16* __restrict__ qkv, const int* __restrict__ padmask,
    float* __restrict__ attn) {
  const int lane  = threadIdx.x & 31;
  const int wave  = threadIdx.x >> 5;
  const int ntile = blockIdx.x * 4 + wave;
  if (ntile >= (SS >> 6)) return;
  const int tm = blockIdx.y << 6;
  const int tn = ntile << 6;
  const int bh = blockIdx.z;
  const int b  = bh >> 2, h = bh & 3;
  const _Float16* qb = qkv + (size_t)b * SS * (3 * CC) + h * DHD;
  const _Float16* kb = qb + CC;
  float* ab = attn + (size_t)bh * SS * SS;
  const int l = lane & 15, hi = lane >> 4;
  const int aOff = hi ? 8 : 0, bOff = hi ? 8 : 0;

  v8f acc[4][4];
#pragma unroll
  for (int a = 0; a < 4; ++a)
#pragma unroll
    for (int c = 0; c < 4; ++c)
#pragma unroll
      for (int r = 0; r < 8; ++r) acc[a][c][r] = 0.0f;

#pragma unroll
  for (int k0 = 0; k0 < DHD; k0 += 32) {
    v16h af[4], bf[4];
#pragma unroll
    for (int mi = 0; mi < 4; ++mi)
      af[mi] = load_a_f16(qb + (size_t)(tm + mi * 16 + l) * (3 * CC) + k0 + aOff);
#pragma unroll
    for (int ni = 0; ni < 4; ++ni)
      bf[ni] = load_b_f16(kb + (size_t)(tn + ni * 16 + l) * (3 * CC) + k0 + bOff);
#pragma unroll
    for (int mi = 0; mi < 4; ++mi)
#pragma unroll
      for (int ni = 0; ni < 4; ++ni)
        acc[mi][ni] = __builtin_amdgcn_wmma_f32_16x16x32_f16(
            false, af[mi], false, bf[ni], (short)0, acc[mi][ni], false, false);
  }

  const int mBase = hi ? 8 : 0;
#pragma unroll
  for (int ni = 0; ni < 4; ++ni) {
    const int n = tn + ni * 16 + l;
    const bool pad = padmask[b * SS + n] != 0;
#pragma unroll
    for (int mi = 0; mi < 4; ++mi) {
#pragma unroll
      for (int r = 0; r < 8; ++r) {
        const int m = tm + mi * 16 + mBase + r;
        float v = acc[mi][ni][r] * 0.125f;     // 1/sqrt(DH)
        if (pad) v = NEGV;
        ab[(size_t)m * SS + n] = v;
      }
    }
  }
}

// ---------------------------------------------------------------------------
// out[b, q, h*DH + d] = softmax_attn(f32) @ V, V pre-transposed to [b,h,DH,S].
// A fragments converted f32 -> f16 in-register. One wave = 64x64 (N=DH=64).
// ---------------------------------------------------------------------------
__global__ __launch_bounds__(32) void gemm_av_kernel(
    const float* __restrict__ attn, const _Float16* __restrict__ vT,
    _Float16* __restrict__ out) {
  const int lane = threadIdx.x & 31;
  const int tm = blockIdx.y << 6;
  const int bh = blockIdx.z;
  const int b  = bh >> 2, h = bh & 3;
  const float* ab = attn + (size_t)bh * SS * SS;
  const _Float16* vb = vT + (size_t)bh * DHD * SS;
  _Float16* ob = out + (size_t)b * SS * CC + h * DHD;
  const int l = lane & 15, hi = lane >> 4;
  const int aOff = hi ? 8 : 0, bOff = hi ? 8 : 0;

  v8f acc[4][4];
#pragma unroll
  for (int a = 0; a < 4; ++a)
#pragma unroll
    for (int c = 0; c < 4; ++c)
#pragma unroll
      for (int r = 0; r < 8; ++r) acc[a][c][r] = 0.0f;

  for (int k0 = 0; k0 < SS; k0 += 32) {
    v16h af[4], bf[4];
#pragma unroll
    for (int mi = 0; mi < 4; ++mi)
      af[mi] = load_a_f32(ab + (size_t)(tm + mi * 16 + l) * SS + k0 + aOff);
#pragma unroll
    for (int ni = 0; ni < 4; ++ni)
      bf[ni] = load_b_f16(vb + (size_t)(ni * 16 + l) * SS + k0 + bOff);
#pragma unroll
    for (int mi = 0; mi < 4; ++mi)
#pragma unroll
      for (int ni = 0; ni < 4; ++ni)
        acc[mi][ni] = __builtin_amdgcn_wmma_f32_16x16x32_f16(
            false, af[mi], false, bf[ni], (short)0, acc[mi][ni], false, false);
  }

  const int mBase = hi ? 8 : 0;
#pragma unroll
  for (int ni = 0; ni < 4; ++ni) {
    const int n = ni * 16 + l;
#pragma unroll
    for (int mi = 0; mi < 4; ++mi)
#pragma unroll
      for (int r = 0; r < 8; ++r) {
        const int m = tm + mi * 16 + mBase + r;
        ob[(size_t)m * CC + n] = (_Float16)acc[mi][ni][r];
      }
  }
}

// ---------------------------------------------------------------------------
// Elementwise / reduction kernels
// ---------------------------------------------------------------------------
__device__ __forceinline__ float block_red_sum(float v, float* sm) {
  const int t = threadIdx.x;
  sm[t] = v; __syncthreads();
  for (int s = blockDim.x >> 1; s > 0; s >>= 1) {
    if (t < s) sm[t] += sm[t + s];
    __syncthreads();
  }
  const float r = sm[0]; __syncthreads();
  return r;
}

__device__ __forceinline__ float block_red_max(float v, float* sm) {
  const int t = threadIdx.x;
  sm[t] = v; __syncthreads();
  for (int s = blockDim.x >> 1; s > 0; s >>= 1) {
    if (t < s) sm[t] = fmaxf(sm[t], sm[t + s]);
    __syncthreads();
  }
  const float r = sm[0]; __syncthreads();
  return r;
}

__global__ void softmax_kernel(float* __restrict__ a) {
  __shared__ float sm[256];
  float* p = a + (size_t)blockIdx.x * SS;
  float vals[4];
  float mx = -1e30f;
#pragma unroll
  for (int j = 0; j < 4; ++j) {
    vals[j] = p[threadIdx.x + 256 * j];
    mx = fmaxf(mx, vals[j]);
  }
  mx = block_red_max(mx, sm);
  float s = 0.0f;
#pragma unroll
  for (int j = 0; j < 4; ++j) { vals[j] = expf(vals[j] - mx); s += vals[j]; }
  s = block_red_sum(s, sm);
  const float inv = 1.0f / s;
#pragma unroll
  for (int j = 0; j < 4; ++j) p[threadIdx.x + 256 * j] = vals[j] * inv;
}

__global__ void ln_kernel(const float* __restrict__ in, const float* __restrict__ g,
                          const float* __restrict__ be, float* __restrict__ out32,
                          _Float16* __restrict__ out16) {
  __shared__ float sm[256];
  const size_t row = blockIdx.x;
  const float v = in[row * CC + threadIdx.x];
  const float mean = block_red_sum(v, sm) * (1.0f / CC);
  const float d = v - mean;
  const float var = block_red_sum(d * d, sm) * (1.0f / CC);
  const float o = d * rsqrtf(var + 1e-5f) * g[threadIdx.x] + be[threadIdx.x];
  out32[row * CC + threadIdx.x] = o;
  out16[row * CC + threadIdx.x] = (_Float16)o;
}

__global__ void transpose_v_kernel(_Float16* __restrict__ vT,
                                   const _Float16* __restrict__ qkv) {
  const int idx = blockIdx.x * 256 + threadIdx.x;  // B*H*DH*S total
  const int s  = idx & (SS - 1);
  const int d  = (idx >> 10) & (DHD - 1);
  const int bh = idx >> 16;
  const int b  = bh >> 2, h = bh & 3;
  vT[(size_t)bh * DHD * SS + (size_t)d * SS + s] =
      qkv[((size_t)b * SS + s) * (3 * CC) + 2 * CC + h * DHD + d];
}

__global__ void cvt16_kernel(_Float16* __restrict__ d, const float* __restrict__ s, int n) {
  const int i = blockIdx.x * 256 + threadIdx.x;
  if (i < n) d[i] = (_Float16)s[i];
}

// W[o][i][t] (f32, [C,C,K]) -> packed f16 [t][o][i]
__global__ void packconv_kernel(_Float16* __restrict__ d, const float* __restrict__ w) {
  const int idx = blockIdx.x * 256 + threadIdx.x;  // C*C*3
  const int t = idx % 3;
  const int rem = idx / 3;
  const int i = rem % CC;
  const int o = rem / CC;
  d[((size_t)t * CC + o) * CC + i] = (_Float16)w[idx];
}

__global__ void pool_kernel(const float* __restrict__ h, const int* __restrict__ pm,
                            float* __restrict__ gf) {
  const int b = blockIdx.x, c = threadIdx.x;
  float acc = 0.0f, cnt = 0.0f;
  for (int s = 0; s < SS; ++s) {
    const bool valid = (pm[b * SS + s] == 0);
    if (valid) {
      cnt += 1.0f;
      acc += h[((size_t)b * SS + s) * CC + c];
    }
  }
  gf[b * CC + c] = acc / fmaxf(cnt, 1.0f);
}

__global__ void heads_kernel(const float* __restrict__ gf,
                             const float* __restrict__ clsW, const float* __restrict__ clsb,
                             const float* __restrict__ projW, const float* __restrict__ projb,
                             float* __restrict__ out) {
  const int b = blockIdx.x, t = threadIdx.x;
  const float* gfb = gf + b * CC;
  if (t < 4) {
    float s = clsb[t];
    for (int k = 0; k < CC; ++k) s += gfb[k] * clsW[t * CC + k];
    out[b * 4 + t] = s;
  } else if (t < 388) {
    const int j = t - 4;
    float s = projb[j];
    for (int k = 0; k < CC; ++k) s += gfb[k] * projW[j * CC + k];
    out[BB * 4 + b * 384 + j] = s;
  }
}

// ---------------------------------------------------------------------------
// Orchestration
// ---------------------------------------------------------------------------
extern "C" void kernel_launch(void* const* d_in, const int* in_sizes, int n_in,
                              void* d_out, int out_size, void* d_ws, size_t ws_size,
                              hipStream_t stream) {
  (void)in_sizes; (void)n_in; (void)out_size; (void)ws_size;

  const float* x       = (const float*)d_in[0];
  const int*   padmask = (const int*)d_in[1];
  const float* tW1[2] = {(const float*)d_in[2], (const float*)d_in[6]};
  const float* tb1[2] = {(const float*)d_in[3], (const float*)d_in[7]};
  const float* tW2[2] = {(const float*)d_in[4], (const float*)d_in[8]};
  const float* tb2[2] = {(const float*)d_in[5], (const float*)d_in[9]};
  const float *Wqkv[2], *bqkv[2], *Wo[2], *bo[2], *g1[2], *be1[2], *g2[2], *be2[2],
              *Wf1[2], *bf1[2], *Wf2[2], *bf2[2];
  for (int i = 0; i < 2; ++i) {
    const int base = 10 + 12 * i;
    Wqkv[i] = (const float*)d_in[base + 0];  bqkv[i] = (const float*)d_in[base + 1];
    Wo[i]   = (const float*)d_in[base + 2];  bo[i]   = (const float*)d_in[base + 3];
    g1[i]   = (const float*)d_in[base + 4];  be1[i]  = (const float*)d_in[base + 5];
    g2[i]   = (const float*)d_in[base + 6];  be2[i]  = (const float*)d_in[base + 7];
    Wf1[i]  = (const float*)d_in[base + 8];  bf1[i]  = (const float*)d_in[base + 9];
    Wf2[i]  = (const float*)d_in[base + 10]; bf2[i]  = (const float*)d_in[base + 11];
  }
  const float* clsW  = (const float*)d_in[34];
  const float* clsb  = (const float*)d_in[35];
  const float* projW = (const float*)d_in[36];
  const float* projb = (const float*)d_in[37];

  float* out     = (float*)d_out;
  float* attnbuf = out + (BB * 4 + BB * 384);  // last_attn region, reused by layer0

  // ---- workspace carve ----
  size_t off = 0;
  auto alloc = [&](size_t bytes) -> void* {
    void* p = (char*)d_ws + off;
    off += (bytes + 255) & ~(size_t)255;
    return p;
  };
  const size_t NACT = (size_t)BB * SS * CC;  // 4,194,304
  float*     h32   = (float*)alloc(NACT * 4);
  float*     t32   = (float*)alloc(NACT * 4);
  _Float16*  h16   = (_Float16*)alloc(NACT * 2);
  _Float16*  y16   = (_Float16*)alloc(NACT * 2);
  _Float16*  vT16  = (_Float16*)alloc(NACT * 2);
  _Float16*  qkv16 = (_Float16*)alloc(NACT * 3 * 2);
  _Float16*  ff16  = (_Float16*)alloc((size_t)BB * SS * DFFD * 2);
  float*     gfeat = (float*)alloc((size_t)BB * CC * 4);
  _Float16 *wc1[2], *wc2[2], *wqkv16[2], *wo16[2], *wf116[2], *wf216[2];
  for (int i = 0; i < 2; ++i) {
    wc1[i]    = (_Float16*)alloc((size_t)3 * CC * CC * 2);
    wc2[i]    = (_Float16*)alloc((size_t)3 * CC * CC * 2);
    wqkv16[i] = (_Float16*)alloc((size_t)3 * CC * CC * 2);
    wo16[i]   = (_Float16*)alloc((size_t)CC * CC * 2);
    wf116[i]  = (_Float16*)alloc((size_t)DFFD * CC * 2);
    wf216[i]  = (_Float16*)alloc((size_t)CC * DFFD * 2);
  }

  // ---- weight conversion / packing ----
  for (int i = 0; i < 2; ++i) {
    packconv_kernel<<<(3 * CC * CC) / 256, 256, 0, stream>>>(wc1[i], tW1[i]);
    packconv_kernel<<<(3 * CC * CC) / 256, 256, 0, stream>>>(wc2[i], tW2[i]);
    cvt16_kernel<<<(3 * CC * CC) / 256, 256, 0, stream>>>(wqkv16[i], Wqkv[i], 3 * CC * CC);
    cvt16_kernel<<<(CC * CC) / 256, 256, 0, stream>>>(wo16[i], Wo[i], CC * CC);
    cvt16_kernel<<<(DFFD * CC) / 256, 256, 0, stream>>>(wf116[i], Wf1[i], DFFD * CC);
    cvt16_kernel<<<(DFFD * CC) / 256, 256, 0, stream>>>(wf216[i], Wf2[i], DFFD * CC);
  }
  cvt16_kernel<<<(int)(NACT / 256), 256, 0, stream>>>(h16, x, (int)NACT);

  const int M = BB * SS;
  auto launch_gemm = [&](const _Float16* A, int lda, const _Float16* W, int ldw,
                         const float* bias, const float* res, int ldres,
                         float* o32, int ldo32, _Float16* o16, int ldo16,
                         int N, int K, int taps, int dil, int mode) {
    GemmWSParams p{A, lda, W, ldw, bias, res, ldres, o32, ldo32, o16, ldo16,
                   M, N, K, taps, dil, (taps > 1) ? (SS - 1) : 0, mode};
    dim3 g((N / 64 + 3) / 4, M / 64, 1);
    gemm_ws_kernel<<<g, 128, 0, stream>>>(p);
  };

  // ---- TCN backbone (kept in [B,S,C] layout; conv = 3 shifted GEMM taps) ----
  launch_gemm(h16, CC, wc1[0], CC, tb1[0], nullptr, 0, nullptr, 0, y16, CC,
              CC, CC, 3, 1, /*relu*/1);
  launch_gemm(y16, CC, wc2[0], CC, tb2[0], x, CC, h32, CC, h16, CC,
              CC, CC, 3, 1, /*relu,+res,relu*/2);
  launch_gemm(h16, CC, wc1[1], CC, tb1[1], nullptr, 0, nullptr, 0, y16, CC,
              CC, CC, 3, 2, 1);
  launch_gemm(y16, CC, wc2[1], CC, tb2[1], h32, CC, h32, CC, h16, CC,
              CC, CC, 3, 2, 2);

  // ---- transformer layers ----
  for (int i = 0; i < 2; ++i) {
    launch_gemm(h16, CC, wqkv16[i], CC, bqkv[i], nullptr, 0, nullptr, 0,
                qkv16, 3 * CC, 3 * CC, CC, 1, 0, 0);
    transpose_v_kernel<<<(int)(NACT / 256), 256, 0, stream>>>(vT16, qkv16);
    gemm_scores_kernel<<<dim3(4, SS / 64, BB * HH), 128, 0, stream>>>(qkv16, padmask, attnbuf);
    softmax_kernel<<<BB * HH * SS, 256, 0, stream>>>(attnbuf);
    gemm_av_kernel<<<dim3(1, SS / 64, BB * HH), 32, 0, stream>>>(attnbuf, vT16, y16);
    launch_gemm(y16, CC, wo16[i], CC, bo[i], h32, CC, t32, CC, nullptr, 0,
                CC, CC, 1, 0, /*+res*/3);
    ln_kernel<<<M, CC, 0, stream>>>(t32, g1[i], be1[i], h32, h16);
    launch_gemm(h16, CC, wf116[i], CC, bf1[i], nullptr, 0, nullptr, 0,
                ff16, DFFD, DFFD, CC, 1, 0, 1);
    launch_gemm(ff16, DFFD, wf216[i], DFFD, bf2[i], h32, CC, t32, CC, nullptr, 0,
                CC, DFFD, 1, 0, 3);
    ln_kernel<<<M, CC, 0, stream>>>(t32, g2[i], be2[i], h32, h16);
  }

  // ---- heads ----
  pool_kernel<<<BB, CC, 0, stream>>>(h32, padmask, gfeat);
  heads_kernel<<<BB, 512, 0, stream>>>(gfeat, clsW, clsb, projW, projb, out);
}
